// MultiHeadAttention_30511447670841
// MI455X (gfx1250) — matrix operand order repaired and checked by reading, not compiled, not run on hardware
//
#include <hip/hip_runtime.h>
#include <hip/hip_bf16.h>
#include <cstdint>

// ---------------------------------------------------------------------------
// Problem constants (from the reference)
// ---------------------------------------------------------------------------
#define B_   4
#define L_   2048
#define H_   512
#define NH_  8
#define HD_  64      // head size
#define ND_  512     // NH_*HD_

typedef _Float16 v16h __attribute__((ext_vector_type(16)));
typedef _Float16 v8h  __attribute__((ext_vector_type(8)));
typedef float    v8f  __attribute__((ext_vector_type(8)));

#define CAT16(lo, hi) __builtin_shufflevector(lo, hi, 0,1,2,3,4,5,6,7,8,9,10,11,12,13,14,15)

// ---------------------------------------------------------------------------
// gfx1250 async global->LDS copy (ASYNCcnt tracked), per ISA 15.18.3 op 98.
// lds_addr: workgroup-relative LDS byte address (low 32 bits of generic ptr).
// ---------------------------------------------------------------------------
__device__ __forceinline__ void async_ld_b128(unsigned lds_addr, const void* gsrc) {
    asm volatile("global_load_async_to_lds_b128 %0, %1, off"
                 :: "v"(lds_addr), "v"(gsrc) : "memory");
}
__device__ __forceinline__ void wait_async_le4() {
    asm volatile("s_wait_asynccnt 0x4" ::: "memory");
}
__device__ __forceinline__ void wait_async_0() {
    asm volatile("s_wait_asynccnt 0x0" ::: "memory");
}

// ---------------------------------------------------------------------------
// Kernel 0a: x (f32) -> x16 (f16), same layout.  4M elements, 8 per thread.
// ---------------------------------------------------------------------------
__global__ __launch_bounds__(256) void k0_cvt_x(
    const float* __restrict__ x, _Float16* __restrict__ x16)
{
    const size_t i = ((size_t)blockIdx.x * 256 + threadIdx.x) * 8;
    float4 f0 = *(const float4*)(x + i);
    float4 f1 = *(const float4*)(x + i + 4);
    v8h h;
    h[0]=(_Float16)f0.x; h[1]=(_Float16)f0.y; h[2]=(_Float16)f0.z; h[3]=(_Float16)f0.w;
    h[4]=(_Float16)f1.x; h[5]=(_Float16)f1.y; h[6]=(_Float16)f1.z; h[7]=(_Float16)f1.w;
    *(v8h*)(x16 + i) = h;
}

// ---------------------------------------------------------------------------
// Kernel 0b: W (f32, K x N row-major) -> Wt (f16, N x K row-major).
// 16x16 LDS tiles, 1024 blocks of 256 threads.
// ---------------------------------------------------------------------------
__global__ __launch_bounds__(256) void k0_wt(
    const float* __restrict__ W, _Float16* __restrict__ Wt)
{
    __shared__ float t[16][17];
    const int tx = threadIdx.x & 15;
    const int ty = threadIdx.x >> 4;
    const int k0 = (blockIdx.x & 31) * 16;   // 512/16 = 32 K-tiles
    const int j0 = (blockIdx.x >> 5) * 16;   // 32 N-tiles
    t[ty][tx] = W[(size_t)(k0 + ty) * ND_ + j0 + tx];
    __syncthreads();
    Wt[(size_t)(j0 + ty) * H_ + k0 + tx] = (_Float16)t[tx][ty];
}

// ---------------------------------------------------------------------------
// Kernel 1:  u = f16( x @ W_u + b_u )   via v_wmma_f32_16x16x32_f16.
//   x16 : (B*L, H) f16,  Wt : (ND, H) f16 (pre-transposed)
// Each wave owns a 16x64 output strip (4 N-tiles), so every A fragment
// feeds 4 WMMAs.  4096 wave-strips, 8 waves per block.
// ---------------------------------------------------------------------------
__global__ __launch_bounds__(256) void k1_proj_gemm(
    const _Float16* __restrict__ x16,
    const _Float16* __restrict__ Wt,
    const float* __restrict__ bU,
    _Float16* __restrict__ u)
{
    const int wid  = threadIdx.x >> 5;
    const int lane = threadIdx.x & 31;
    const int grp  = lane >> 4;
    const int lr   = lane & 15;

    const int wt = blockIdx.x * 8 + wid;        // 0 .. 4095
    const int m0 = (wt & 511) * 16;             // 512 M-tiles
    const int j0 = (wt >> 9) * 64;              // 8 N-strips of 64 cols

    const _Float16* xrow = x16 + (size_t)(m0 + lr) * H_;
    const _Float16* wrow = Wt  + (size_t)(j0 + lr) * H_;

    v8f acc[4] = {v8f{}, v8f{}, v8f{}, v8f{}};
    #pragma unroll 2
    for (int k0 = 0; k0 < H_; k0 += 32) {
        // A fragment (16x32 f16): lanes 0-15: K=[0..15]; 16-31: K=[16..31].
        v8h x0 = *(const v8h*)(xrow + k0 + grp * 16);
        v8h x1 = *(const v8h*)(xrow + k0 + grp * 16 + 8);
        v16h a = CAT16(x0, x1);
        #pragma unroll
        for (int jt = 0; jt < 4; ++jt) {
            // B fragment (32x16): lane c holds col j0+jt*16+c, K slice by grp.
            v16h bf = *(const v16h*)(wrow + (size_t)jt * 16 * H_ + k0 + grp * 16);
            acc[jt] = __builtin_amdgcn_wmma_f32_16x16x32_f16(
                        false, a, false, bf, (short)0, acc[jt], false, false);
        }
    }

    #pragma unroll
    for (int jt = 0; jt < 4; ++jt) {
        const int jc = j0 + jt * 16 + lr;
        const float bias = bU[jc];
        #pragma unroll
        for (int r = 0; r < 8; ++r) {
            const int row = m0 + grp * 8 + r;
            u[(size_t)row * ND_ + jc] = (_Float16)(acc[jt][r] + bias);
        }
    }
}

// ---------------------------------------------------------------------------
// Kernel 2: flash-style single pass over the Gram matrix A = U U^T per head.
//   Block = 4 waves, shares one (b, head); covers 128 rows (2 row-blocks per
//   wave).  B-operand tiles (u rows) are double-buffered in LDS via
//   global_load_async_to_lds_b128 + s_wait_asynccnt.  Per-lane online
//   max/sum with a SINGLE v_exp_f32 per element; cross-lane merge at the end.
// ---------------------------------------------------------------------------
#define CH_     64                 // k-rows per LDS chunk (4 wmma k-tiles)
#define PITCH_  144                // 128B row + 16B pad (bank-conflict free)

__global__ __launch_bounds__(128) void k2_gram_softmax(
    const _Float16* __restrict__ u,
    float* __restrict__ part_m,
    float* __restrict__ part_s)
{
    __shared__ __align__(16) unsigned char smem[2 * CH_ * PITCH_];

    const int tid  = threadIdx.x;
    const int wid  = tid >> 5;
    const int lane = tid & 31;
    const int grp  = lane >> 4;
    const int lr   = lane & 15;

    const int bb = blockIdx.x;          // 0 .. 511
    const int bn = bb >> 4;             // (b,n) index, 0 .. 31
    const int n  = bn & 7;
    const int b  = bn >> 3;
    const int r0 = (bb & 15) * 128 + wid * 32;   // wave's first row

    const _Float16* ub = u + (size_t)b * L_ * ND_ + n * HD_;   // + row*ND_

    // ---- persistent A fragments: 2 row-blocks x (16x64) ----
    v16h a[2][2];
    #pragma unroll
    for (int rbp = 0; rbp < 2; ++rbp) {
        const _Float16* p = ub + (size_t)(r0 + rbp * 16 + lr) * ND_;
        v8h q0 = *(const v8h*)(p + grp * 8);
        v8h q1 = *(const v8h*)(p + 16 + grp * 8);
        v8h q2 = *(const v8h*)(p + 32 + grp * 8);
        v8h q3 = *(const v8h*)(p + 48 + grp * 8);
        a[rbp][0] = CAT16(q0, q1);
        a[rbp][1] = CAT16(q2, q3);
    }

    // per-lane online softmax state (rows x this lane's column slice)
    float mr[2][8], sr[2][8];
    #pragma unroll
    for (int rbp = 0; rbp < 2; ++rbp)
        #pragma unroll
        for (int r = 0; r < 8; ++r) { mr[rbp][r] = -__builtin_inff(); sr[rbp][r] = 0.0f; }

    const unsigned lds_base = (unsigned)(uintptr_t)(void*)smem;

    // cooperative async copy of one 64-row chunk (8KB) into buf (chunk&1)
    auto issue_chunk = [&](int chunk) {
        const int k0 = chunk * CH_;
        const unsigned bo = lds_base + (unsigned)(chunk & 1) * (CH_ * PITCH_);
        #pragma unroll
        for (int i = 0; i < 4; ++i) {
            const int unit = i * 128 + tid;        // 512 x 16B units
            const int row  = unit >> 3;
            const int c16  = unit & 7;
            const unsigned dst = bo + (unsigned)row * PITCH_ + (unsigned)c16 * 16;
            const void* src = (const char*)(ub + (size_t)(k0 + row) * ND_) + c16 * 16;
            async_ld_b128(dst, src);
        }
    };

    issue_chunk(0);
    for (int c = 0; c < L_ / CH_; ++c) {
        if (c + 1 < L_ / CH_) { issue_chunk(c + 1); wait_async_le4(); }
        else                  { wait_async_0(); }
        __syncthreads();

        const unsigned char* bo = smem + (c & 1) * (CH_ * PITCH_);
        #pragma unroll
        for (int t = 0; t < CH_ / 16; ++t) {
            // B operand (64x16 = u_cols^T) from LDS: lane c holds column c,
            // i.e. u row (chunk*64 + t*16 + c), d-slice by half-wave.
            const unsigned char* prow = bo + (unsigned)(t * 16 + lr) * PITCH_ + grp * 16;
            v8h q0 = *(const v8h*)(prow);
            v8h q1 = *(const v8h*)(prow + 32);
            v8h q2 = *(const v8h*)(prow + 64);
            v8h q3 = *(const v8h*)(prow + 96);
            v16h b0 = CAT16(q0, q1);        // d0..31
            v16h b1 = CAT16(q2, q3);        // d32..63

            #pragma unroll
            for (int rbp = 0; rbp < 2; ++rbp) {
                v8f acc = {};
                acc = __builtin_amdgcn_wmma_f32_16x16x32_f16(
                        false, a[rbp][0], false, b0, (short)0, acc, false, false);
                acc = __builtin_amdgcn_wmma_f32_16x16x32_f16(
                        false, a[rbp][1], false, b1, (short)0, acc, false, false);
                // online (m,s) update, ONE exp per element:
                //   d = a - m; e = exp(-|d|)
                //   d<=0:  s += e              (m unchanged)
                //   d>0 :  s = s*e + 1; m = a
                #pragma unroll
                for (int r = 0; r < 8; ++r) {
                    const float av = acc[r];
                    const float m_ = mr[rbp][r];
                    const float d  = av - m_;
                    const float e  = __expf(-__builtin_fabsf(d));
                    const bool  gt = d > 0.0f;
                    sr[rbp][r] = fmaf(sr[rbp][r], gt ? e : 1.0f, gt ? 1.0f : e);
                    mr[rbp][r] = gt ? av : m_;
                }
            }
        }
        __syncthreads();
    }

    // ---- cross-lane merge (associative (m,s) combine) + writeout ----
    #pragma unroll
    for (int rbp = 0; rbp < 2; ++rbp) {
        #pragma unroll
        for (int r = 0; r < 8; ++r) {
            float M = mr[rbp][r];
            M = fmaxf(M, __shfl_xor(M, 1));
            M = fmaxf(M, __shfl_xor(M, 2));
            M = fmaxf(M, __shfl_xor(M, 4));
            M = fmaxf(M, __shfl_xor(M, 8));
            float w = sr[rbp][r] * __expf(mr[rbp][r] - M);
            w += __shfl_xor(w, 1);
            w += __shfl_xor(w, 2);
            w += __shfl_xor(w, 4);
            w += __shfl_xor(w, 8);
            if (lr == 0) {
                const int row = r0 + rbp * 16 + grp * 8 + r;
                const size_t idx = ((size_t)(b * NH_ + n)) * L_ + row;
                part_m[idx] = M;
                part_s[idx] = w;
            }
        }
    }
}

// ---------------------------------------------------------------------------
// Kernel 3: one block per batch.  Merge (m,s) over heads, global max +
// denominator via block reduction, alpha in LDS, attended = alpha^T x.
// ---------------------------------------------------------------------------
__global__ __launch_bounds__(1024) void k3_finalize(
    const float* __restrict__ x,
    const float* __restrict__ part_m,
    const float* __restrict__ part_s,
    float* __restrict__ out)
{
    __shared__ float lm[L_];
    __shared__ float ls[L_];
    __shared__ float red[1024];

    const int b   = blockIdx.x;
    const int tid = threadIdx.x;

    for (int i = tid; i < L_; i += 1024) {
        float mm = -__builtin_inff();
        #pragma unroll
        for (int n = 0; n < NH_; ++n)
            mm = fmaxf(mm, part_m[((size_t)(b * NH_ + n)) * L_ + i]);
        float ss = 0.0f;
        #pragma unroll
        for (int n = 0; n < NH_; ++n) {
            const size_t idx = ((size_t)(b * NH_ + n)) * L_ + i;
            ss += part_s[idx] * __expf(part_m[idx] - mm);
        }
        lm[i] = mm;
        ls[i] = ss;
    }
    __syncthreads();

    float lmax = -__builtin_inff();
    for (int i = tid; i < L_; i += 1024) lmax = fmaxf(lmax, lm[i]);
    red[tid] = lmax;
    __syncthreads();
    #pragma unroll
    for (int off = 512; off > 0; off >>= 1) {
        if (tid < off) red[tid] = fmaxf(red[tid], red[tid + off]);
        __syncthreads();
    }
    const float M = red[0];
    __syncthreads();

    float lsum = 0.0f;
    for (int i = tid; i < L_; i += 1024) {
        const float wv = ls[i] * __expf(lm[i] - M);
        ls[i] = wv;
        lsum += wv;
    }
    __syncthreads();
    red[tid] = lsum;
    __syncthreads();
    #pragma unroll
    for (int off = 512; off > 0; off >>= 1) {
        if (tid < off) red[tid] += red[tid + off];
        __syncthreads();
    }
    const float inv_denom = 1.0f / red[0];
    __syncthreads();

    for (int i = tid; i < L_; i += 1024) ls[i] *= inv_denom;   // alpha
    __syncthreads();

    const int h    = tid & (H_ - 1);
    const int part = tid >> 9;
    float acc = 0.0f;
    const float* xb = x + (size_t)b * L_ * H_;
    for (int l = part * (L_ / 2); l < (part + 1) * (L_ / 2); ++l)
        acc = fmaf(ls[l], xb[(size_t)l * H_ + h], acc);
    red[tid] = acc;
    __syncthreads();
    if (tid < H_) out[b * H_ + tid] = red[tid] + red[tid + H_];
}

// ---------------------------------------------------------------------------
// launch
// ---------------------------------------------------------------------------
extern "C" void kernel_launch(void* const* d_in, const int* in_sizes, int n_in,
                              void* d_out, int out_size, void* d_ws, size_t ws_size,
                              hipStream_t stream) {
    const float* x  = (const float*)d_in[0];
    const float* W  = (const float*)d_in[1];
    const float* bU = (const float*)d_in[2];
    float* out = (float*)d_out;

    // workspace layout (bytes)
    char* ws = (char*)d_ws;
    _Float16* u   = (_Float16*)(ws);                        //  8 MiB
    _Float16* x16 = (_Float16*)(ws + 8388608);              //  8 MiB
    _Float16* Wt  = (_Float16*)(ws + 16777216);             // 512 KiB
    float* part_m = (float*)   (ws + 17301504);             // 256 KiB
    float* part_s = (float*)   (ws + 17563648);             // 256 KiB

    // operand conversion
    k0_cvt_x<<<2048, 256, 0, stream>>>(x, x16);             // 4M f32 -> f16
    k0_wt  <<<1024, 256, 0, stream>>>(W, Wt);               // transpose+cvt W

    // K1: 4096 wave-strips (16x64 each), 8 waves per 256-thread block
    k1_proj_gemm<<<512, 256, 0, stream>>>(x16, Wt, bU, u);

    // K2: 512 blocks x 4 waves; each block owns 128 rows of one (b,head)
    k2_gram_softmax<<<512, 128, 0, stream>>>(u, part_m, part_s);

    // K3: one block per batch
    k3_finalize<<<B_, 1024, 0, stream>>>(x, part_m, part_s, out);
}
